// MonarchTransformer_52304111730700
// MI455X (gfx1250) — matrix operand order repaired and checked by reading, not compiled, not run on hardware
//
#include <hip/hip_runtime.h>

typedef __bf16 bf16_t;
typedef __attribute__((ext_vector_type(16))) __bf16 v16bf;
typedef __attribute__((ext_vector_type(8)))  __bf16 v8bf;
typedef __attribute__((ext_vector_type(8)))  float  v8f;

#define NBATCH 4
#define SEQLEN 4096
#define NTOK   (NBATCH * SEQLEN)   // 16384
#define DM     1024

// ---------- fragment layout helpers ----------
// 16-bit A-matrix 16x32 lane layout (ISA 7.12.2):
// lane<16 : row M=lane,    K = {0..7, 16..23}
// lane>=16: row M=lane-16, K = {8..15, 24..31}
__device__ __forceinline__ int a_kmap(int lane, int j) {
  int half = lane >> 4;
  return (j < 8) ? (half * 8 + j) : (16 + half * 8 + (j - 8));
}

__device__ __forceinline__ v8f wmma_bf16(v16bf a, v16bf b, v8f c) {
  return __builtin_amdgcn_wmma_f32_16x16x32_bf16(false, a, false, b, (short)0, c,
                                                 false, false);
}

__device__ __forceinline__ float selu_f(float x) {
  const float scale = 1.0507009873554805f, alpha = 1.6732632423543772f;
  return scale * (x > 0.0f ? x : alpha * (__expf(x) - 1.0f));
}

// Scatter one C tile (rows M = cRow+j, this lane's column) into an A-fragment
// region whose M equals C's row and whose K equals `kidx` (0..31).
// Inverse of a_kmap: dest lane = 16*((kidx>>3)&1) + (cRow+j), elem = (kidx&7)+8*(kidx>>4).
__device__ __forceinline__ void c_scatter_afrag(bf16_t* frag, const v8f& c, int lane,
                                                int kidx) {
  int cRow = (lane >> 4) * 8;
  int lbase = (((kidx >> 3) & 1) << 4) + cRow;
  int jpp = (kidx & 7) + ((kidx >> 4) << 3);
#pragma unroll
  for (int j = 0; j < 8; ++j) frag[(lbase + j) * 16 + jpp] = (bf16_t)c[j];
}

// ---------- TDM 1-D copy global -> LDS (CDNA5 Tensor Data Mover) ----------
__device__ __forceinline__ void tdm_copy_to_lds(void* ldsDst, const void* gsrc,
                                                unsigned nbytes /* mult of 8, <=512KB */) {
#if defined(__gfx1250__)
  typedef __attribute__((ext_vector_type(4))) unsigned int v4u;
  typedef __attribute__((ext_vector_type(8))) int v8i;
  typedef __attribute__((ext_vector_type(4))) int v4i;
  unsigned n8 = nbytes >> 3;
  unsigned lds = (unsigned)(unsigned long long)ldsDst;
  unsigned long long ga = (unsigned long long)gsrc;
  v4u g0;
  g0[0] = 1u;                                    // count=1 (user descriptor)
  g0[1] = lds;                                   // lds_addr
  g0[2] = (unsigned)(ga & 0xffffffffu);          // global_addr[31:0]
  g0[3] = (unsigned)((ga >> 32) & 0x1ffffffu) | (2u << 30);  // addr[56:32] | type=2
  v8i g1;
  g1[0] = (int)(3u << 16);                       // data_size = 3 (8B units)
  g1[1] = (int)((n8 & 0xffffu) << 16);           // tensor_dim0[15:0]
  g1[2] = (int)(((n8 >> 16) & 0xffffu) | (1u << 16));  // dim0[31:16] | tensor_dim1=1
  g1[3] = (int)((n8 & 0xffffu) << 16);           // tile_dim0
  g1[4] = 1;                                     // tile_dim1 = 1
  g1[5] = (int)n8;                               // tensor_dim0_stride[31:0]
  g1[6] = 0;
  g1[7] = 0;
  v4i z4 = {0, 0, 0, 0};
#if __clang_major__ >= 23
  v8i z8 = {0, 0, 0, 0, 0, 0, 0, 0};
  __builtin_amdgcn_tensor_load_to_lds(g0, g1, z4, z4, z8, 0);
#else
  __builtin_amdgcn_tensor_load_to_lds(g0, g1, z4, z4, 0);
#endif
  __builtin_amdgcn_s_wait_tensorcnt(0);
#else
  (void)ldsDst; (void)gsrc; (void)nbytes;
#endif
}

// ---------- pack kernels ----------
// Generic f32 [outer][K=32][N] -> bf16 B-fragment layout.
__global__ void pack_b_kernel(const float* __restrict__ src, bf16_t* __restrict__ dst,
                              int nOuter, int ntiles, int Ntot, int Kstride,
                              int outerStride) {
  int tid = blockIdx.x * blockDim.x + threadIdx.x;
  int total = nOuter * ntiles * 512;
  if (tid >= total) return;
  int j = tid & 15;
  int lane = (tid >> 4) & 31;
  int nt = (tid >> 9) % ntiles;
  int outer = tid / (512 * ntiles);
  int k = ((lane >> 4) << 3) + (j & 7) + ((j >> 3) << 4);
  int n = nt * 16 + (lane & 15);
  float v = (n < Ntot) ? src[outer * outerStride + k * Kstride + n] : 0.0f;
  dst[tid] = (bf16_t)v;
}

// data (f32, [tok][1024]) -> A-fragment pack for Monarch stage 1:
// apack[((tile*32 + b)*32 + lane)*16 + j] = data[(tile*16 + lane%16)*1024 + a_kmap(lane,j)*32 + b]
__global__ void pack_data_a(const float* __restrict__ src, bf16_t* __restrict__ dst) {
  int tid = blockIdx.x * blockDim.x + threadIdx.x;  // (tile, b, lane)
  int lane = tid & 31;
  int b = (tid >> 5) & 31;
  int tile = tid >> 10;
  if (tile >= NTOK / 16) return;
  int row = lane & 15;
  const float* s = src + (size_t)(tile * 16 + row) * DM + b;
  v16bf o;
#pragma unroll
  for (int j = 0; j < 16; ++j) o[j] = (bf16_t)s[(size_t)a_kmap(lane, j) * 32];
  *(v16bf*)(dst + (size_t)tid * 16) = o;
}

// ---------- kernel A1: fused dual-Monarch (qkv + ff), SELU ----------
// grid 256 x 128 threads (4 waves); wave handles 16 tokens.
__global__ void monarch_front(const bf16_t* __restrict__ data_apack,
                              const bf16_t* __restrict__ qkvL_pack,
                              const bf16_t* __restrict__ qkvR_pack,
                              const bf16_t* __restrict__ ffL_pack,
                              const bf16_t* __restrict__ ffR_pack,
                              const float* __restrict__ qkv_bias,
                              const float* __restrict__ ff_bias,
                              bf16_t* __restrict__ qkv_out,   // [NTOK][1152]
                              float* __restrict__ ffo_out) {  // [NTOK][1024]
  extern __shared__ char smemraw[];
  const int wave = threadIdx.x >> 5;
  const int lane = threadIdx.x & 31;
  // per-wave A-fragment staging regions (next-stage layout)
  bf16_t* tT  = (bf16_t*)(smemraw) + wave * (24 * 512);                 // 24 p-frags
  bf16_t* t2T = (bf16_t*)(smemraw + 4 * 24 * 512 * 2) + wave * (16 * 512);  // 16 h-frags
  const int tile = blockIdx.x * 4 + wave;
  const int g0 = tile * 16;
  const int cRow = (lane >> 4) * 8;
  const int col = lane & 15;

  // ----- stage 1 (shared A between qkv and ff monarchs), K = m = 32 -----
  for (int b = 0; b < 32; ++b) {
    v16bf a = *(const v16bf*)(data_apack + ((size_t)(tile * 32 + b) * 32 + lane) * 16);
    __builtin_prefetch(data_apack + ((size_t)(tile * 32 + b + 1) * 32 + lane) * 16, 0, 1);
    v16bf bq0 = *(const v16bf*)(qkvL_pack + ((b * 2 + 0) * 32 + lane) * 16);
    v16bf bq1 = *(const v16bf*)(qkvL_pack + ((b * 2 + 1) * 32 + lane) * 16);
    v16bf bff = *(const v16bf*)(ffL_pack + (b * 32 + lane) * 16);
    v8f c0 = {}, c1 = {}, c2 = {};
    c0 = wmma_bf16(a, bq0, c0);
    c1 = wmma_bf16(a, bq1, c1);
    c2 = wmma_bf16(a, bff, c2);
    c_scatter_afrag(tT + col * 512, c0, lane, b);
    if (col < 8) c_scatter_afrag(tT + (16 + col) * 512, c1, lane, b);
    c_scatter_afrag(t2T + col * 512, c2, lane, b);
  }
  __syncthreads();

  // ----- stage 2 qkv: K = b = 32, per p three 16-wide N tiles (q=48) -----
  for (int p = 0; p < 24; ++p) {
    v16bf a = *(const v16bf*)(tT + p * 512 + lane * 16);
    for (int nt = 0; nt < 3; ++nt) {
      v16bf bb = *(const v16bf*)(qkvR_pack + ((p * 3 + nt) * 32 + lane) * 16);
      v8f c = {};
      c = wmma_bf16(a, bb, c);
      int q = nt * 16 + col;
      float bias = qkv_bias[p * 48 + q];
#pragma unroll
      for (int j = 0; j < 8; ++j) {
        int g = g0 + cRow + j;
        qkv_out[(size_t)g * 1152 + p * 48 + q] = (bf16_t)selu_f(c[j] + bias);
      }
    }
  }

  // ----- stage 2 ff: per h four 16-wide N tiles (v=64) -----
  for (int h = 0; h < 16; ++h) {
    v16bf a = *(const v16bf*)(t2T + h * 512 + lane * 16);
    for (int nt = 0; nt < 4; ++nt) {
      v16bf bb = *(const v16bf*)(ffR_pack + ((h * 4 + nt) * 32 + lane) * 16);
      v8f c = {};
      c = wmma_bf16(a, bb, c);
      int v = nt * 16 + col;
      float bias = ff_bias[h * 64 + v];
#pragma unroll
      for (int j = 0; j < 8; ++j) {
        int g = g0 + cRow + j;
        ffo_out[(size_t)g * 1024 + h * 64 + v] = selu_f(c[j] + bias);
      }
    }
  }
}

// ---------- rope math: l2norm + 3x3 section rotation + frequency RoPE on a 64-vec ----------
__device__ __forceinline__ void rot_rope64(float* x, const float* __restrict__ rotations,
                                           const float* __restrict__ locations,
                                           const float* __restrict__ freqs, int g) {
  float ss = 0.f;
#pragma unroll
  for (int i = 0; i < 64; ++i) ss += x[i] * x[i];
  float inv = 1.0f / (sqrtf(ss) + 1e-4f);
#pragma unroll
  for (int i = 0; i < 64; ++i) x[i] *= inv;
  float r0 = rotations[g * 9 + 0], r1 = rotations[g * 9 + 1], r2 = rotations[g * 9 + 2];
  float r3 = rotations[g * 9 + 3], r4 = rotations[g * 9 + 4], r5 = rotations[g * 9 + 5];
  float r6 = rotations[g * 9 + 6], r7 = rotations[g * 9 + 7], r8 = rotations[g * 9 + 8];
#pragma unroll
  for (int d = 0; d < 16; ++d) {
    float s0 = x[d], s1 = x[16 + d], s2 = x[32 + d];
    x[d]      = r0 * s0 + r1 * s1 + r2 * s2;
    x[16 + d] = r3 * s0 + r4 * s1 + r5 * s2;
    x[32 + d] = r6 * s0 + r7 * s1 + r8 * s2;
  }
  float loc[3] = {locations[g * 3], locations[g * 3 + 1], locations[g * 3 + 2]};
#pragma unroll
  for (int d = 0; d < 30; ++d) {
    float fm = freqs[d / 3] * loc[d % 3];
    float s = __sinf(fm), c = __cosf(fm);
    float a0 = x[2 * d], a1 = x[2 * d + 1];
    x[2 * d]     = c * a0 + s * a1;
    x[2 * d + 1] = -s * a0 + c * a1;
  }
}

// Write a 64-vec (A-row M = rowIdx of some 16-row tile) into packed A-frag layout:
// dst is the 1024-element (2 kfrags x 32 lanes x 16) region of that tile.
__device__ __forceinline__ void write_afrag64(bf16_t* dst, const float* x, int rowIdx) {
#pragma unroll
  for (int f = 0; f < 2; ++f)
#pragma unroll
    for (int hl = 0; hl < 2; ++hl) {
      int l = hl * 16 + rowIdx;
      v16bf o;
#pragma unroll
      for (int j = 0; j < 16; ++j) o[j] = (bf16_t)x[f * 32 + a_kmap(l, j)];
      *(v16bf*)(dst + (f * 32 + l) * 16) = o;
    }
}

// ---------- kernel A2q: l2norm + rope per (token, head); emits A-frag pack ----------
__global__ void rope_q(const bf16_t* __restrict__ qkv, const float* __restrict__ locations,
                       const float* __restrict__ rotations, const float* __restrict__ freqs,
                       bf16_t* __restrict__ qh_apack) {
  int tid = blockIdx.x * blockDim.x + threadIdx.x;
  int h = tid & 15;
  int g = tid >> 4;
  if (g >= NTOK) return;
  float x[64];
  const bf16_t* src = qkv + (size_t)g * 1152 + h * 64;
#pragma unroll
  for (int i = 0; i < 64; ++i) x[i] = (float)src[i];
  rot_rope64(x, rotations, locations, freqs, g);
  write_afrag64(qh_apack + (size_t)g * 1024, x, h);  // A rows = heads of this token
}

// ---------- kernel A2kv: K -> A-frag pack, V -> B-frag pack ----------
__global__ void rope_kv(const bf16_t* __restrict__ qkv, const float* __restrict__ locations,
                        const float* __restrict__ rotations, const float* __restrict__ freqs,
                        bf16_t* __restrict__ k_apack, bf16_t* __restrict__ v_bpack) {
  int g = blockIdx.x * blockDim.x + threadIdx.x;
  if (g >= NTOK) return;
  {
    float x[64];
    const bf16_t* src = qkv + (size_t)g * 1152 + 1024;
#pragma unroll
    for (int i = 0; i < 64; ++i) x[i] = (float)src[i];
    rot_rope64(x, rotations, locations, freqs, g);
    write_afrag64(k_apack + (size_t)(g >> 4) * 1024, x, g & 15);  // A rows = tokens of tile
  }
  {
    float x[64];
    const bf16_t* src = qkv + (size_t)g * 1152 + 1088;
    float ss = 0.f;
#pragma unroll
    for (int i = 0; i < 64; ++i) { x[i] = (float)src[i]; ss += x[i] * x[i]; }
    float inv = 1.0f / (sqrtf(ss) + 1e-4f);
    // B-frag pack: token g is K-position g%32 of K-step g/32.
    int jj = (g & 7) + (((g >> 4) & 1) << 3), hK = (g >> 3) & 1;
    bf16_t* dst = v_bpack + (size_t)(g >> 5) * 2048;  // 4 vtiles * 512
#pragma unroll
    for (int vt = 0; vt < 4; ++vt)
#pragma unroll
      for (int c = 0; c < 16; ++c)
        dst[(vt * 32 + hK * 16 + c) * 16 + jj] = (bf16_t)(x[vt * 16 + c] * inv);
  }
}

// ---------- kernel B: wk = exp(K@W) and partial kv = wk^T @ V per 128-token chunk ----------
// grid 128 x 256 threads (8 waves).
__global__ void kv_accum(const bf16_t* __restrict__ k_apack, const bf16_t* __restrict__ v_bpack,
                         const bf16_t* __restrict__ w_pack, float* __restrict__ kv_partial) {
  extern __shared__ char smemraw[];
  bf16_t* wkA = (bf16_t*)smemraw;                    // [ks4][rt16][32][16] = 64KB
  bf16_t* kA  = (bf16_t*)(smemraw + 65536);          // [tt8][f2][32][16]  = 16KB
  const int wave = threadIdx.x >> 5;
  const int lane = threadIdx.x & 31;
  const int col = lane & 15;
  const int cRow = (lane >> 4) * 8;

  if (wave == 0)  // TDM: async 16KB pull of this chunk's packed-K fragments
    tdm_copy_to_lds(kA, k_apack + (size_t)blockIdx.x * 8192, 16384);
  __syncthreads();

  // B fragments of W for this wave's two n-tiles
  v16bf bw[2][2];
#pragma unroll
  for (int t = 0; t < 2; ++t)
#pragma unroll
    for (int f = 0; f < 2; ++f)
      bw[t][f] = *(const v16bf*)(w_pack + ((f * 16 + wave * 2 + t) * 32 + lane) * 16);

  // phase 1: wk tiles; key trick: C(tok,r) -> A(M=r,K=tok) transpose collapses to
  // one contiguous 16B LDS store per tile (dest lane == source lane).
  for (int tt = 0; tt < 8; ++tt) {
    v16bf a0 = *(const v16bf*)(kA + ((tt * 2 + 0) * 32 + lane) * 16);
    v16bf a1 = *(const v16bf*)(kA + ((tt * 2 + 1) * 32 + lane) * 16);
#pragma unroll
    for (int t = 0; t < 2; ++t) {
      int nt = wave * 2 + t;
      v8f c = {};
      c = wmma_bf16(a0, bw[t][0], c);
      c = wmma_bf16(a1, bw[t][1], c);
      v8bf o;
#pragma unroll
      for (int j = 0; j < 8; ++j) o[j] = (bf16_t)__expf(c[j]);
      *(v8bf*)(wkA + ((tt >> 1) * 16 + nt) * 512 + lane * 16 + (tt & 1) * 8) = o;
    }
  }
  __syncthreads();

  // phase 2: kv += wk^T @ V ; wave owns r-tiles {2w,2w+1} x v-tiles 0..3
  v8f acc[2][4] = {};
  for (int kk = 0; kk < 4; ++kk) {
    v16bf aA[2];
#pragma unroll
    for (int t = 0; t < 2; ++t)
      aA[t] = *(const v16bf*)(wkA + ((kk * 16 + wave * 2 + t) * 32 + lane) * 16);
#pragma unroll
    for (int vt = 0; vt < 4; ++vt) {
      v16bf bb = *(const v16bf*)(v_bpack + (size_t)(blockIdx.x * 4 + kk) * 2048 +
                                 (vt * 32 + lane) * 16);
      acc[0][vt] = wmma_bf16(aA[0], bb, acc[0][vt]);
      acc[1][vt] = wmma_bf16(aA[1], bb, acc[1][vt]);
    }
  }
  const size_t base = (size_t)blockIdx.x * (256 * 64);
#pragma unroll
  for (int t = 0; t < 2; ++t)
#pragma unroll
    for (int vt = 0; vt < 4; ++vt)
#pragma unroll
      for (int j = 0; j < 8; ++j) {
        int r = (wave * 2 + t) * 16 + cRow + j;
        int v = vt * 16 + col;
        kv_partial[base + r * 64 + v] = acc[t][vt][j];
      }
}

// ---------- kernel C: deterministic reduce over 32 chunks -> kv in B-fragment pack ----------
__global__ void kv_reduce(const float* __restrict__ kv_partial, bf16_t* __restrict__ kv_pack) {
  int tid = blockIdx.x * blockDim.x + threadIdx.x;
  int j = tid & 15, lane = (tid >> 4) & 31, nt = (tid >> 9) & 3, kk = (tid >> 11) & 7;
  int batch = tid >> 14;
  if (batch >= NBATCH) return;
  int r = kk * 32 + ((lane >> 4) * 8 + (j & 7) + ((j >> 3) << 4));
  int v = nt * 16 + (lane & 15);
  float s = 0.f;
  for (int ch = 0; ch < 32; ++ch)
    s += kv_partial[((size_t)(batch * 32 + ch) * 256 + r) * 64 + v];
  kv_pack[(size_t)batch * 16384 + ((kk * 4 + nt) * 32 + lane) * 16 + j] = (bf16_t)s;
}

// ---------- kernel D: wq = exp(Qh@W), O = wq@kv, +FFO, LayerNorm. One wave per token ----------
__global__ void attn_out(const bf16_t* __restrict__ qh_apack, const bf16_t* __restrict__ w_pack,
                         const bf16_t* __restrict__ kv_pack, const float* __restrict__ ffo,
                         const float* __restrict__ ln_w, const float* __restrict__ ln_b,
                         float* __restrict__ out) {
  extern __shared__ char smemraw[];
  const int wave = threadIdx.x >> 5;
  const int lane = threadIdx.x & 31;
  bf16_t* wqA = (bf16_t*)smemraw + wave * 4096;       // per-wave [ks8][32][16] = 8KB
  bf16_t* wL  = (bf16_t*)(smemraw + 65536);           // w_pack copy, 16KB
  bf16_t* kvL = (bf16_t*)(smemraw + 65536 + 16384);   // batch kv copy, 32KB
  const int g = blockIdx.x * 8 + wave;
  const int batch = g >> 12;
  const int cRow = (lane >> 4) * 8;
  const int col = lane & 15;

  if (wave == 0) {  // TDM: stage W and this batch's kv through LDS
    tdm_copy_to_lds(wL, w_pack, 16384);
    tdm_copy_to_lds(kvL, kv_pack + (size_t)batch * 16384, 32768);
  }
  __syncthreads();

  // A fragments (M = heads) straight from the packed global layout
  v16bf a0 = *(const v16bf*)(qh_apack + (size_t)g * 1024 + lane * 16);
  v16bf a1 = *(const v16bf*)(qh_apack + (size_t)g * 1024 + (32 + lane) * 16);

  // wq = exp(Qh @ W): 16 n-tiles; scatter into O-stage A-fragment layout
  for (int nt = 0; nt < 16; ++nt) {
    v16bf b0 = *(const v16bf*)(wL + ((0 * 16 + nt) * 32 + lane) * 16);
    v16bf b1 = *(const v16bf*)(wL + ((1 * 16 + nt) * 32 + lane) * 16);
    v8f c = {};
    c = wmma_bf16(a0, b0, c);
    c = wmma_bf16(a1, b1, c);
    v8f e;
#pragma unroll
    for (int j = 0; j < 8; ++j) e[j] = __expf(c[j]);
    c_scatter_afrag(wqA + (nt >> 1) * 512, e, lane, (nt & 1) * 16 + col);
  }
  __syncthreads();

  // O = wq @ kv : 8 K-steps x 4 v-tiles
  v8f acc[4] = {};
  for (int kk = 0; kk < 8; ++kk) {
    v16bf a = *(const v16bf*)(wqA + kk * 512 + lane * 16);
#pragma unroll
    for (int vt = 0; vt < 4; ++vt) {
      v16bf bb = *(const v16bf*)(kvL + ((kk * 4 + vt) * 32 + lane) * 16);
      acc[vt] = wmma_bf16(a, bb, acc[vt]);
    }
  }

  // epilogue: x = O + FFO, wave32 LayerNorm
  float xv[4][8];
  float sum = 0.f, sumsq = 0.f;
#pragma unroll
  for (int vt = 0; vt < 4; ++vt)
#pragma unroll
    for (int j = 0; j < 8; ++j) {
      int h = cRow + j, v = vt * 16 + col;
      float x = acc[vt][j] + ffo[(size_t)g * 1024 + h * 64 + v];
      xv[vt][j] = x;
      sum += x;
      sumsq += x * x;
    }
#pragma unroll
  for (int off = 16; off > 0; off >>= 1) {
    sum += __shfl_xor(sum, off, 32);
    sumsq += __shfl_xor(sumsq, off, 32);
  }
  float mean = sum * (1.0f / 1024.0f);
  float var = sumsq * (1.0f / 1024.0f) - mean * mean;
  float rstd = rsqrtf(var + 1e-5f);
#pragma unroll
  for (int vt = 0; vt < 4; ++vt)
#pragma unroll
    for (int j = 0; j < 8; ++j) {
      int h = cRow + j, v = vt * 16 + col;
      int idx = h * 64 + v;
      out[(size_t)g * 1024 + idx] = (xv[vt][j] - mean) * rstd * ln_w[idx] + ln_b[idx];
    }
}

// ---------- host launch ----------
extern "C" void kernel_launch(void* const* d_in, const int* in_sizes, int n_in,
                              void* d_out, int out_size, void* d_ws, size_t ws_size,
                              hipStream_t stream) {
  (void)in_sizes; (void)n_in; (void)out_size; (void)ws_size;
  const float* data      = (const float*)d_in[0];
  const float* locations = (const float*)d_in[1];
  const float* rotations = (const float*)d_in[2];
  const float* qkv_L     = (const float*)d_in[3];
  const float* qkv_R     = (const float*)d_in[4];
  const float* qkv_bias  = (const float*)d_in[5];
  const float* ff_L      = (const float*)d_in[6];
  const float* ff_R      = (const float*)d_in[7];
  const float* ff_bias   = (const float*)d_in[8];
  const float* W         = (const float*)d_in[9];
  const float* freqs     = (const float*)d_in[10];
  const float* ln_w      = (const float*)d_in[11];
  const float* ln_b      = (const float*)d_in[12];
  float* out = (float*)d_out;

  char* ws = (char*)d_ws;
  size_t off = 0;
  auto take = [&](size_t bytes) -> char* {
    char* p = ws + off;
    off += (bytes + 255) & ~(size_t)255;
    return p;
  };
  bf16_t* qkv_ws     = (bf16_t*)take((size_t)NTOK * 1152 * 2);
  float*  ffo        = (float*) take((size_t)NTOK * 1024 * 4);
  bf16_t* data_apack = (bf16_t*)take((size_t)NTOK * 1024 * 2);
  bf16_t* qh_apack   = (bf16_t*)take((size_t)NTOK * 1024 * 2);
  bf16_t* k_apack    = (bf16_t*)take((size_t)(NTOK / 16) * 1024 * 2);
  bf16_t* v_bpack    = (bf16_t*)take((size_t)(NTOK / 32) * 2048 * 2);
  bf16_t* qkvL_pack  = (bf16_t*)take(32768 * 2);
  bf16_t* qkvR_pack  = (bf16_t*)take(36864 * 2);
  bf16_t* ffL_pack   = (bf16_t*)take(16384 * 2);
  bf16_t* ffR_pack   = (bf16_t*)take(32768 * 2);
  bf16_t* w_pack     = (bf16_t*)take(16384 * 2);
  float*  kv_partial = (float*) take((size_t)128 * 256 * 64 * 4);
  bf16_t* kv_pack    = (bf16_t*)take((size_t)NBATCH * 16384 * 2);

  pack_b_kernel<<<128, 256, 0, stream>>>(qkv_L, qkvL_pack, 32, 2, 24, 24, 768);
  pack_b_kernel<<<144, 256, 0, stream>>>(qkv_R, qkvR_pack, 24, 3, 48, 48, 1536);
  pack_b_kernel<<<64, 256, 0, stream>>>(ff_L, ffL_pack, 32, 1, 16, 16, 512);
  pack_b_kernel<<<128, 256, 0, stream>>>(ff_R, ffR_pack, 16, 4, 64, 64, 2048);
  pack_b_kernel<<<64, 256, 0, stream>>>(W, w_pack, 2, 16, 256, 256, 8192);
  pack_data_a<<<4096, 256, 0, stream>>>(data, data_apack);

  hipFuncSetAttribute((const void*)monarch_front,
                      hipFuncAttributeMaxDynamicSharedMemorySize, 163840);
  hipFuncSetAttribute((const void*)kv_accum,
                      hipFuncAttributeMaxDynamicSharedMemorySize, 81920);
  hipFuncSetAttribute((const void*)attn_out,
                      hipFuncAttributeMaxDynamicSharedMemorySize, 114688);

  monarch_front<<<256, 128, 163840, stream>>>(data_apack, qkvL_pack, qkvR_pack, ffL_pack,
                                              ffR_pack, qkv_bias, ff_bias, qkv_ws, ffo);
  rope_q<<<1024, 256, 0, stream>>>(qkv_ws, locations, rotations, freqs, qh_apack);
  rope_kv<<<64, 256, 0, stream>>>(qkv_ws, locations, rotations, freqs, k_apack, v_bpack);
  kv_accum<<<128, 256, 81920, stream>>>(k_apack, v_bpack, w_pack, kv_partial);
  kv_reduce<<<256, 256, 0, stream>>>(kv_partial, kv_pack);
  attn_out<<<2048, 256, 114688, stream>>>(qh_apack, w_pack, kv_pack, ffo, ln_w, ln_b, out);
}
